// Head_86088324481600
// MI455X (gfx1250) — matrix operand order repaired and checked by reading, not compiled, not run on hardware
//
#include <hip/hip_runtime.h>
#include <hip/hip_bf16.h>
#include <math.h>

typedef __attribute__((ext_vector_type(16))) __bf16 v16bf;
typedef __attribute__((ext_vector_type(8)))  float  v8f;

#define TQ   2048
#define CDIM 1024
#define HD   64

__device__ __forceinline__ v8f wmma_bf16(v16bf a, v16bf b, v8f c) {
  return __builtin_amdgcn_wmma_f32_16x16x32_bf16(false, a, false, b, (short)0, c, false, false);
}

__device__ __forceinline__ float red_max16(float v) {
#pragma unroll
  for (int m = 1; m <= 8; m <<= 1) v = fmaxf(v, __shfl_xor(v, m, 32));
  return v;
}
__device__ __forceinline__ float red_sum16(float v) {
#pragma unroll
  for (int m = 1; m <= 8; m <<= 1) v += __shfl_xor(v, m, 32);
  return v;
}

// Build 16-bit A-operand (16x32, M=l&15) from an f32 global row.
// e in 0..15 -> K = half*16 + e
__device__ __forceinline__ v16bf load_A_f32(const float* rowptr, int half) {
  union { v16bf v; __bf16 h[16]; } u;
  const float* p0 = rowptr + half * 16;
  const float* p1 = rowptr + half * 16 + 8;
#pragma unroll
  for (int e = 0; e < 8; ++e) { u.h[e] = (__bf16)p0[e]; u.h[8 + e] = (__bf16)p1[e]; }
  return u.v;
}

// Load 16 contiguous bf16 from 32B-aligned LDS (B-operand: N=l&15, K=half*16+e).
__device__ __forceinline__ v16bf lds_load_16bf(const __bf16* p) {
  union { v16bf v; uint4 q[2]; } u;
  const uint4* s = (const uint4*)p;
  u.q[0] = s[0];
  u.q[1] = s[1];
  return u.v;
}

// Convert 8 contiguous f32 -> 8 bf16 and store as one 16B LDS write.
__device__ __forceinline__ void cvt_store8(const float* src, __bf16* dst) {
  union { __bf16 h[8]; uint4 u; } t;
#pragma unroll
  for (int e = 0; e < 8; ++e) t.h[e] = (__bf16)src[e];
  *(uint4*)dst = t.u;
}

// ---------------- Kernel 1: fused QKV projection ----------------
// y = x @ W^T for Wq/Wk/Wv, x viewed as [B*T, C]. 256 thr = 8 waves, 128 rows/block.
__global__ __launch_bounds__(256) void qkv_kernel(
    const float* __restrict__ x,
    const float* __restrict__ Wk, const float* __restrict__ Wq, const float* __restrict__ Wv,
    float* __restrict__ qout, float* __restrict__ kout, float* __restrict__ vout) {
  __shared__ __attribute__((aligned(32))) __bf16 wlds[3 * 64 * 32];  // [mat][n][k], 12 KB

  const int tid  = threadIdx.x;
  const int wave = tid >> 5;
  const int lane = tid & 31;
  const int half = lane >> 4;
  const int l15  = lane & 15;
  const int rowbase = blockIdx.x * 128 + wave * 16;

  const float* Ws[3] = {Wq, Wk, Wv};
  v8f acc[3][4] = {};

  for (int kc = 0; kc < CDIM; kc += 32) {
    // Stage Wq/Wk/Wv[:, kc:kc+32] into LDS as bf16 (768 groups of 8, 3 per thread).
#pragma unroll
    for (int i = 0; i < 3; ++i) {
      int g   = tid + i * 256;
      int mat = g >> 8;
      int gg  = g & 255;
      int n   = gg >> 2;
      int ko  = (gg & 3) * 8;
      const float* src = Ws[mat] + (size_t)n * CDIM + kc + ko;
      cvt_store8(src, &wlds[mat * 2048 + n * 32 + ko]);
      if (kc + 32 < CDIM) __builtin_prefetch(src + 32, 0, 0);  // next K-slice
    }
    __syncthreads();

    const float* xrow = x + (size_t)(rowbase + l15) * CDIM + kc;
    v16bf a = load_A_f32(xrow, half);
    if (kc + 32 < CDIM) __builtin_prefetch(xrow + 32, 0, 0);

#pragma unroll
    for (int m = 0; m < 3; ++m) {
      // Batch the 4 B tiles into distinct registers so DS loads overlap WMMAs
      v16bf bt[4];
#pragma unroll
      for (int nt = 0; nt < 4; ++nt)
        bt[nt] = lds_load_16bf(&wlds[m * 2048 + (nt * 16 + l15) * 32 + half * 16]);
#pragma unroll
      for (int nt = 0; nt < 4; ++nt)
        acc[m][nt] = wmma_bf16(a, bt[nt], acc[m][nt]);
    }
    __syncthreads();
  }

  float* outs[3] = {qout, kout, vout};
#pragma unroll
  for (int m = 0; m < 3; ++m)
#pragma unroll
    for (int nt = 0; nt < 4; ++nt)
#pragma unroll
      for (int r = 0; r < 8; ++r) {
        int row = rowbase + half * 8 + r;
        outs[m][(size_t)row * HD + nt * 16 + l15] = acc[m][nt][r];
      }
}

// ---------------- Kernel 2: causal flash attention ----------------
// 8 waves x 16 query rows = 128 rows/block; key blocks of 32 staged in LDS.
__global__ __launch_bounds__(256) void attn_kernel(
    const float* __restrict__ q, const float* __restrict__ k, const float* __restrict__ v,
    float* __restrict__ out) {
  __shared__ __attribute__((aligned(32))) __bf16 kbuf[32 * 64];     // [key][dim]   4 KB
  __shared__ __attribute__((aligned(32))) __bf16 vbufT[64 * 32];    // [dim][key]   4 KB
  __shared__ __attribute__((aligned(32))) __bf16 pbuf[8 * 16 * 32]; // per-wave P   8 KB

  const int tid  = threadIdx.x;
  const int wave = tid >> 5;
  const int lane = tid & 31;
  const int half = lane >> 4;
  const int l15  = lane & 15;
  const int b     = blockIdx.x >> 4;          // 16 row-tiles of 128 per batch
  const int qbase = (blockIdx.x & 15) * 128;
  const int qtile = qbase + wave * 16;

  const float* qb = q + (size_t)b * TQ * HD;
  const float* kg = k + (size_t)b * TQ * HD;
  const float* vg = v + (size_t)b * TQ * HD;

  // q A-operands for head-dim K=0..31 and K=32..63 (loaded once).
  const float* qrow = qb + (size_t)(qtile + l15) * HD;
  v16bf qa0 = load_A_f32(qrow, half);
  v16bf qa1 = load_A_f32(qrow + 32, half);

  float mrow[8], lrow[8];
  v8f o[4] = {};
#pragma unroll
  for (int r = 0; r < 8; ++r) { mrow[r] = -INFINITY; lrow[r] = 0.f; }

  const int kbEnd = (qbase + 127) >> 5;
  for (int kb = 0; kb <= kbEnd; ++kb) {
    const int key0 = kb * 32;
    {   // cooperative staging: 8 dims of one key per thread
      int key = tid >> 3;
      int d0  = (tid & 7) * 8;
      const float* ks = kg + (size_t)(key0 + key) * HD + d0;
      cvt_store8(ks, &kbuf[key * 64 + d0]);
      const float* vs = vg + (size_t)(key0 + key) * HD + d0;
#pragma unroll
      for (int e = 0; e < 8; ++e) vbufT[(d0 + e) * 32 + key] = (__bf16)vs[e];
      if (key0 + 32 < TQ) {
        __builtin_prefetch(ks + 32 * HD, 0, 0);
        __builtin_prefetch(vs + 32 * HD, 0, 0);
      }
    }
    __syncthreads();

    if (key0 <= qtile + 15) {   // wave-uniform: EXEC all-ones inside (WMMA requirement)
      // S = q @ k^T : two 16x16 D tiles, K(head dim)=64 -> 2 accumulating WMMAs each.
      // Load all 4 B tiles first so DS traffic overlaps the WMMAs.
      v16bf kb0 = lds_load_16bf(&kbuf[l15 * 64 + half * 16]);
      v16bf kb1 = lds_load_16bf(&kbuf[l15 * 64 + 32 + half * 16]);
      v16bf kb2 = lds_load_16bf(&kbuf[(16 + l15) * 64 + half * 16]);
      v16bf kb3 = lds_load_16bf(&kbuf[(16 + l15) * 64 + 32 + half * 16]);
      v8f s0 = {}, s1 = {};
      s0 = wmma_bf16(qa0, kb0, s0);
      s1 = wmma_bf16(qa0, kb2, s1);
      s0 = wmma_bf16(qa1, kb1, s0);
      s1 = wmma_bf16(qa1, kb3, s1);

      const float sc = 0.125f;  // HEAD_SIZE^-0.5
      __bf16* pw = &pbuf[wave * 512];
#pragma unroll
      for (int r = 0; r < 8; ++r) {
        int row = qtile + half * 8 + r;
        float e0 = (key0 + l15      <= row) ? s0[r] * sc : -INFINITY;
        float e1 = (key0 + 16 + l15 <= row) ? s1[r] * sc : -INFINITY;
        float rm    = red_max16(fmaxf(e0, e1));
        float mnew  = fmaxf(mrow[r], rm);
        float scale = __expf(mrow[r] - mnew);
        float p0 = __expf(e0 - mnew);
        float p1 = __expf(e1 - mnew);
        lrow[r] = lrow[r] * scale + red_sum16(p0 + p1);
        mrow[r] = mnew;
#pragma unroll
        for (int t = 0; t < 4; ++t) o[t][r] *= scale;
        int prow = half * 8 + r;
        pw[prow * 32 + l15]      = (__bf16)p0;
        pw[prow * 32 + 16 + l15] = (__bf16)p1;
      }
      // wave-private LDS round-trip: D-layout -> A-layout; fence DS ops
      __asm volatile("s_wait_dscnt 0" ::: "memory");
      v16bf pa;
      {
        union { v16bf v; uint4 qv[2]; } u;
        u.qv[0] = *(const uint4*)&pw[l15 * 32 + half * 16];       // K = half*16 + 0..7
        u.qv[1] = *(const uint4*)&pw[l15 * 32 + half * 16 + 8];  // K = half*16 + 8..15
        pa = u.v;
      }
      // O += P @ V : four 16x16 D tiles over head dim (batched B loads)
      v16bf bv[4];
#pragma unroll
      for (int t = 0; t < 4; ++t)
        bv[t] = lds_load_16bf(&vbufT[(t * 16 + l15) * 32 + half * 16]);
#pragma unroll
      for (int t = 0; t < 4; ++t)
        o[t] = wmma_bf16(pa, bv[t], o[t]);
    }
    __syncthreads();
  }

  float* ob = out + (size_t)b * TQ * HD;
#pragma unroll
  for (int t = 0; t < 4; ++t)
#pragma unroll
    for (int r = 0; r < 8; ++r) {
      int row = qtile + half * 8 + r;
      ob[(size_t)row * HD + t * 16 + l15] = o[t][r] / lrow[r];
    }
}

extern "C" void kernel_launch(void* const* d_in, const int* in_sizes, int n_in,
                              void* d_out, int out_size, void* d_ws, size_t ws_size,
                              hipStream_t stream) {
  const float* x  = (const float*)d_in[0];
  const float* Wk = (const float*)d_in[1];
  const float* Wq = (const float*)d_in[2];
  const float* Wv = (const float*)d_in[3];

  const int B = in_sizes[0] / (TQ * CDIM);   // 8
  const size_t BT = (size_t)B * TQ;

  float* ctx  = (float*)d_out;               // [B*T*64]
  float* kout = ctx + BT * HD;               // [B*T*64]
  float* vout = kout + BT * HD;              // [B*T*64]
  float* qws  = (float*)d_ws;                // q scratch: B*T*64 f32 = 4 MB

  qkv_kernel<<<dim3((unsigned)(BT / 128)), 256, 0, stream>>>(x, Wk, Wq, Wv, qws, kout, vout);
  attn_kernel<<<dim3((unsigned)(B * (TQ / 128))), 256, 0, stream>>>(qws, kout, vout, ctx);
}
